// MultiHeadLatentAttn_65240553226461
// MI455X (gfx1250) — matrix operand order repaired and checked by reading, not compiled, not run on hardware
//
#include <hip/hip_runtime.h>
#include <hip/hip_bf16.h>
#include <math.h>

// ---------------------------------------------------------------------------
// MLA forward for MI455X (gfx1250): bf16 WMMA (v_wmma_f32_16x16x32_bf16),
// wave32, fused flash-attention, double-buffered async global->LDS staging in
// the GEMMs. V is kept transposed [B,H,DV,S] so PV B-fragments are contiguous
// 32B global loads (no LDS staging / column reads). fp32 accumulate.
// ---------------------------------------------------------------------------

typedef __bf16 bf16;
typedef bf16  v16bf __attribute__((ext_vector_type(16)));
typedef bf16  v8bf  __attribute__((ext_vector_type(8)));
typedef bf16  v4bf  __attribute__((ext_vector_type(4)));
typedef float v8f   __attribute__((ext_vector_type(8)));
typedef int   v4i   __attribute__((ext_vector_type(4)));

#define WMMA_BF16(a, b, c) \
  __builtin_amdgcn_wmma_f32_16x16x32_bf16(false, (a), false, (b), (short)0, (c), false, false)

#if __has_builtin(__builtin_amdgcn_global_load_async_to_lds_b128) && \
    __has_builtin(__builtin_amdgcn_s_wait_asynccnt)
#define HAVE_ASYNC 1
#else
#define HAVE_ASYNC 0
#endif

typedef __attribute__((address_space(1))) v4i* gv4i_t;   // global int4*
typedef __attribute__((address_space(3))) v4i* lv4i_t;   // LDS int4*

static constexpr int Bc = 8, Sc = 512, Dc = 2048, Hc = 16;
static constexpr int DOWNc = 512, UPc = 2048, Rc = 512;
static constexpr int BSc = Bc * Sc;          // 4096 rows for all GEMMs
static constexpr int DQK = 640;              // VHD(128) + R(512)
static constexpr int DV  = 128;

__device__ __forceinline__ v8f v8f_zero() {
  v8f z;
#pragma unroll
  for (int i = 0; i < 8; ++i) z[i] = 0.0f;
  return z;
}
__device__ __forceinline__ v16bf cat8(v8bf a, v8bf b) {
  return __builtin_shufflevector(a, b, 0, 1, 2, 3, 4, 5, 6, 7,
                                 8, 9, 10, 11, 12, 13, 14, 15);
}

// ---------------------------------------------------------------------------
// fp32 -> bf16 bulk convert (4 elements / thread)
// ---------------------------------------------------------------------------
__global__ __launch_bounds__(256) void cvt_bf16_kernel(
    const float* __restrict__ src, bf16* __restrict__ dst, long n) {
  long i = ((long)blockIdx.x * blockDim.x + threadIdx.x) * 4;
  if (i >= n) return;
  float4 v = *(const float4*)(src + i);
  v4bf o;
  o[0] = (bf16)v.x; o[1] = (bf16)v.y; o[2] = (bf16)v.z; o[3] = (bf16)v.w;
  *(v4bf*)(dst + i) = o;
}

// ---------------------------------------------------------------------------
// GEMM: C[M,N] = A[M,K]*W[K,N] + bias  (A,W bf16; C fp32 or bf16 by template)
// Block tile 128x128, 8 waves (2x4), wave tile 64x32, K-step 32.
// Double-buffered LDS: tile k+1 staged (A via async DMA, W transposed) while
// the 8 WMMAs consume tile k; ASYNCcnt waited only at iteration boundary.
// ---------------------------------------------------------------------------
template <int BF16OUT>
__global__ __launch_bounds__(256) void gemm_bf16_kernel(
    const bf16* __restrict__ A, const bf16* __restrict__ W,
    const float* __restrict__ bias, float* __restrict__ Cf,
    bf16* __restrict__ Cb, int M, int N, int K) {
  __shared__ bf16 As[2][128][40];   // [m][k], 80B row stride (16B aligned)
  __shared__ bf16 Bs[2][128][40];   // transposed: [n][k]

  const int tid  = threadIdx.x;
  const int wave = tid >> 5;
  const int lane = tid & 31;
  const int half = lane >> 4;
  const int l16  = lane & 15;
  const int wm   = wave >> 2;       // 0..1  -> 64-row slice
  const int wn   = wave & 3;        // 0..3  -> 32-col slice
  const int bm   = blockIdx.y * 128;
  const int bn   = blockIdx.x * 128;

  auto stage = [&](int buf, int k0) {
    // A tile 128x32 bf16 (8KB): 512 x 16B transfers, 2/thread
#pragma unroll
    for (int i = 0; i < 2; ++i) {
      int id = tid + i * 256;
      int r = id >> 2, c = id & 3;
#if HAVE_ASYNC
      __builtin_amdgcn_global_load_async_to_lds_b128(
          (gv4i_t)(A + (size_t)(bm + r) * K + k0 + c * 8),
          (lv4i_t)&As[buf][r][c * 8], 0, 0);
#else
      *(v8bf*)&As[buf][r][c * 8] =
          *(const v8bf*)(A + (size_t)(bm + r) * K + k0 + c * 8);
#endif
    }
    // W tile 32x128 -> Bs[n][k] transposed
#pragma unroll
    for (int i = 0; i < 2; ++i) {
      int id = tid + i * 256;            // 512 v8bf
      int r = id >> 4, c8 = id & 15;     // r: k row, c8: 8-col group
      v8bf w = *(const v8bf*)(W + (size_t)(k0 + r) * N + bn + c8 * 8);
#pragma unroll
      for (int j = 0; j < 8; ++j) Bs[buf][c8 * 8 + j][r] = w[j];
    }
  };

  v8f acc[4][2];
#pragma unroll
  for (int i = 0; i < 4; ++i)
#pragma unroll
    for (int j = 0; j < 2; ++j) acc[i][j] = v8f_zero();

  stage(0, 0);
#if HAVE_ASYNC
  __builtin_amdgcn_s_wait_asynccnt(0);
#endif
  __syncthreads();

  int buf = 0;
  for (int k0 = 0; k0 < K; k0 += 32) {
    if (k0 + 32 < K) stage(buf ^ 1, k0 + 32);   // prefetch next tile

    v16bf afr[4], bfr[2];
#pragma unroll
    for (int i = 0; i < 4; ++i) {
      int m = wm * 64 + i * 16 + l16;
      const v8bf* ap = (const v8bf*)&As[buf][m][half * 16];
      afr[i] = cat8(ap[0], ap[1]);       // k = half*16 + e
    }
#pragma unroll
    for (int j = 0; j < 2; ++j) {
      int n = wn * 32 + j * 16 + l16;
      const v8bf* bp = (const v8bf*)&Bs[buf][n][half * 16];
      bfr[j] = cat8(bp[0], bp[1]);       // k = half*16 + e
    }
#pragma unroll
    for (int i = 0; i < 4; ++i)
#pragma unroll
      for (int j = 0; j < 2; ++j)
        acc[i][j] = WMMA_BF16(afr[i], bfr[j], acc[i][j]);

#if HAVE_ASYNC
    __builtin_amdgcn_s_wait_asynccnt(0);
#endif
    __syncthreads();
    buf ^= 1;
  }

  // Epilogue: C layout: lane col = l16, VGPR g -> row g + 8*half.
#pragma unroll
  for (int j = 0; j < 2; ++j) {
    int n = bn + wn * 32 + j * 16 + l16;
    float bv = bias[n];
#pragma unroll
    for (int i = 0; i < 4; ++i) {
      int mbase = bm + wm * 64 + i * 16 + half * 8;
#pragma unroll
      for (int g = 0; g < 8; ++g) {
        float r = acc[i][j][g] + bv;
        size_t off = (size_t)(mbase + g) * N + n;
        if constexpr (BF16OUT) Cb[off] = (bf16)r;
        else                   Cf[off] = r;
      }
    }
  }
}

// ---------------------------------------------------------------------------
// Pack [B,S,H*128] bf16 -> [B,H,S,ld] bf16 at column offset `off` (8 elem/thr)
// ---------------------------------------------------------------------------
__global__ __launch_bounds__(256) void pack_heads_kernel(
    const bf16* __restrict__ src, bf16* __restrict__ dst,
    long total8, int ld, int off) {
  long idx = (long)blockIdx.x * blockDim.x + threadIdx.x;
  if (idx >= total8) return;
  long e0 = idx * 8;
  int d  = (int)(e0 & 127);
  long t = e0 >> 7;
  int h  = (int)(t & (Hc - 1));
  long t2 = t >> 4;
  int s  = (int)(t2 & (Sc - 1));
  int b  = (int)(t2 >> 9);
  *(v8bf*)(dst + (((size_t)b * Hc + h) * Sc + s) * ld + off + d) =
      *(const v8bf*)(src + e0);
}

// ---------------------------------------------------------------------------
// Pack V transposed: [B,S,H*128] bf16 -> Vt[B,H,128,S] bf16.
// Each thread gathers 8 seq positions of one (b,h,d) and stores 16B vector.
// ---------------------------------------------------------------------------
__global__ __launch_bounds__(256) void pack_vt_kernel(
    const bf16* __restrict__ src, bf16* __restrict__ dst, long total8) {
  long idx = (long)blockIdx.x * blockDim.x + threadIdx.x;
  if (idx >= total8) return;
  int s8 = (int)(idx & 63);            // 64 groups of 8 along S
  long t  = idx >> 6;
  int d  = (int)(t & (DV - 1));
  long t2 = t >> 7;
  int h  = (int)(t2 & (Hc - 1));
  int b  = (int)(t2 >> 4);
  int s0 = s8 * 8;
  v8bf v;
#pragma unroll
  for (int j = 0; j < 8; ++j)
    v[j] = src[(size_t)(b * Sc + s0 + j) * (Hc * DV) + h * DV + d];
  *(v8bf*)(dst + (((size_t)b * Hc + h) * DV + d) * Sc + s0) = v;
}

// ---------------------------------------------------------------------------
// RoPE on q_r: [B,S,H*512] bf16 -> Q[B,H,S,640] bf16 at offset 128.
// theta(s,p) = s * BASE^(-2p/512); interleaved pairs.
// ---------------------------------------------------------------------------
__global__ __launch_bounds__(256) void rope_q_kernel(
    const bf16* __restrict__ src, bf16* __restrict__ dst, long total) {
  long idx = (long)blockIdx.x * blockDim.x + threadIdx.x;
  if (idx >= total) return;
  int p  = (int)(idx & 255);
  long t = idx >> 8;
  int h  = (int)(t % Hc);
  long t2 = t / Hc;
  int s  = (int)(t2 % Sc);
  int b  = (int)(t2 / Sc);
  float freq = __expf(-0.0359778963f * (float)p);  // ln(1e4)/256
  float sn, cs;
  __sincosf((float)s * freq, &sn, &cs);
  size_t sb = ((size_t)b * Sc + s) * (Rc * Hc) + (size_t)h * Rc;
  float q1 = (float)src[sb + 2 * p], q2 = (float)src[sb + 2 * p + 1];
  size_t db = (((size_t)b * Hc + h) * Sc + s) * DQK + 128;
  dst[db + 2 * p]     = (bf16)(q1 * cs - q2 * sn);
  dst[db + 2 * p + 1] = (bf16)(q2 * cs + q1 * sn);
}

// RoPE on k_r: [B,S,512] bf16 -> broadcast to all H heads of K at offset 128.
__global__ __launch_bounds__(256) void rope_k_kernel(
    const bf16* __restrict__ src, bf16* __restrict__ dst, long total) {
  long idx = (long)blockIdx.x * blockDim.x + threadIdx.x;
  if (idx >= total) return;
  int p  = (int)(idx & 255);
  long t = idx >> 8;
  int s  = (int)(t % Sc);
  int b  = (int)(t / Sc);
  float freq = __expf(-0.0359778963f * (float)p);
  float sn, cs;
  __sincosf((float)s * freq, &sn, &cs);
  size_t sb = ((size_t)b * Sc + s) * Rc;
  float q1 = (float)src[sb + 2 * p], q2 = (float)src[sb + 2 * p + 1];
  float o1 = q1 * cs - q2 * sn;
  float o2 = q2 * cs + q1 * sn;
#pragma unroll
  for (int h = 0; h < Hc; ++h) {
    size_t db = (((size_t)b * Hc + h) * Sc + s) * DQK + 128;
    dst[db + 2 * p]     = (bf16)o1;
    dst[db + 2 * p + 1] = (bf16)o2;
  }
}

// ---------------------------------------------------------------------------
// Fused attention: grid (B*H, S/16), 128 threads = 4 waves.
// Wave w: score keys [w*16,w*16+16); PV output cols [w*32,w*32+32).
// Online softmax over 64-key blocks. Q/K/V fragments load straight from
// global (V transposed so PV B-frags are contiguous 32B loads); only the
// P tile and softmax state live in LDS. Output bf16 [B,S,H*128].
// ---------------------------------------------------------------------------
__global__ __launch_bounds__(128) void mla_attn_kernel(
    const bf16* __restrict__ Qg, const bf16* __restrict__ Kg,
    const bf16* __restrict__ Vt, bf16* __restrict__ Og) {
  __shared__ bf16  Ps[16][72];        // P tile (16q x 64k), 144B row stride
  __shared__ float red_m[4][16], red_l[4][16];
  __shared__ float rowm[16], rowl[16], alpha_s[16];

  const int bh   = blockIdx.x;              // b*H + h
  const int qb   = blockIdx.y;              // 16-row q tile
  const int h    = bh & (Hc - 1);
  const int b    = bh >> 4;
  const int tid  = threadIdx.x;
  const int wave = tid >> 5;
  const int lane = tid & 31;
  const int half = lane >> 4;
  const int l16  = lane & 15;

  const bf16* Qt = Qg + (size_t)bh * Sc * DQK + (size_t)qb * 16 * DQK;
  const bf16* Kh = Kg + (size_t)bh * Sc * DQK;
  const bf16* Vh = Vt + (size_t)bh * DV * Sc;
  const float sc = 0.0294627825f;           // 1/(sqrt(128)+sqrt(512))

  v8f o0 = v8f_zero(), o1 = v8f_zero();
  if (tid < 16) { rowm[tid] = -1e30f; rowl[tid] = 0.0f; }
  __syncthreads();

  for (int kb = 0; kb < Sc; kb += 64) {
    // ---- scores: 16q x 16k per wave, K-dim = 640 in chunks of 32 ----
    v8f s = v8f_zero();
    const bf16* Kt = Kh + (size_t)(kb + wave * 16) * DQK;
    for (int d0 = 0; d0 < DQK; d0 += 32) {
      const v8bf* qp = (const v8bf*)(Qt + (size_t)l16 * DQK + d0 + half * 16);
      v16bf a = cat8(qp[0], qp[1]);        // d = d0+half*16+e
      v16bf bq = *(const v16bf*)(Kt + (size_t)l16 * DQK + d0 + half * 16);
      s = WMMA_BF16(a, bq, s);
    }

    // ---- per-wave row max (rows half*8+g, reduce over 16 cols) ----
    float sv[8], pm[8];
#pragma unroll
    for (int g = 0; g < 8; ++g) { sv[g] = s[g] * sc; pm[g] = sv[g]; }
#pragma unroll
    for (int off = 8; off >= 1; off >>= 1)
#pragma unroll
      for (int g = 0; g < 8; ++g) pm[g] = fmaxf(pm[g], __shfl_xor(pm[g], off));
    if (l16 == 0)
#pragma unroll
      for (int g = 0; g < 8; ++g) red_m[wave][half * 8 + g] = pm[g];
    __syncthreads();                                   // (A)

    if (tid < 16) {
      float nm = rowm[tid];
      for (int w = 0; w < 4; ++w) nm = fmaxf(nm, red_m[w][tid]);
      float al = __expf(rowm[tid] - nm);
      rowl[tid] *= al;
      rowm[tid] = nm;
      alpha_s[tid] = al;
    }
    __syncthreads();                                   // (B)

    // ---- p = exp(s - m), stage P in LDS, partial row sums ----
    float ps[8];
#pragma unroll
    for (int g = 0; g < 8; ++g) {
      int row = half * 8 + g;
      float pv = __expf(sv[g] - rowm[row]);
      Ps[row][wave * 16 + l16] = (bf16)pv;
      ps[g] = pv;
    }
#pragma unroll
    for (int off = 8; off >= 1; off >>= 1)
#pragma unroll
      for (int g = 0; g < 8; ++g) ps[g] += __shfl_xor(ps[g], off);
    if (l16 == 0)
#pragma unroll
      for (int g = 0; g < 8; ++g) red_l[wave][half * 8 + g] = ps[g];
    __syncthreads();                                   // (C) Ps+red_l ready

    if (tid < 16)
      rowl[tid] += red_l[0][tid] + red_l[1][tid] + red_l[2][tid] + red_l[3][tid];

    // ---- rescale O and accumulate P*V (wave cols wave*32 + {0,16}) ----
#pragma unroll
    for (int g = 0; g < 8; ++g) {
      float al = alpha_s[half * 8 + g];
      o0[g] *= al; o1[g] *= al;
    }
#pragma unroll
    for (int kk = 0; kk < 2; ++kk) {
      const v8bf* pp = (const v8bf*)&Ps[l16][kk * 32 + half * 16];
      v16bf pa = cat8(pp[0], pp[1]);       // key = kk*32 + half*16+e
      // V^T rows are contiguous over keys: one 32B load per fragment
      const bf16* vtb = Vh + (size_t)(wave * 32 + l16) * Sc + kb + kk * 32 + half * 16;
      v16bf vb0 = *(const v16bf*)vtb;
      v16bf vb1 = *(const v16bf*)(vtb + (size_t)16 * Sc);
      o0 = WMMA_BF16(pa, vb0, o0);
      o1 = WMMA_BF16(pa, vb1, o1);
    }
    // next iteration's Ps writes are fenced by barriers (A)+(B)
  }
  __syncthreads();                         // rowl final values visible

  // ---- normalize and store O as [B,S,H*128] bf16 ----
#pragma unroll
  for (int g = 0; g < 8; ++g) {
    int row = half * 8 + g;
    float inv = 1.0f / rowl[row];
    size_t base = ((size_t)b * Sc + qb * 16 + row) * (Hc * DV) + (size_t)h * DV + wave * 32;
    Og[base + l16]      = (bf16)(o0[g] * inv);
    Og[base + 16 + l16] = (bf16)(o1[g] * inv);
  }
}

// ---------------------------------------------------------------------------
extern "C" void kernel_launch(void* const* d_in, const int* in_sizes, int n_in,
                              void* d_out, int out_size, void* d_ws, size_t ws_size,
                              hipStream_t stream) {
  (void)in_sizes; (void)n_in; (void)out_size; (void)ws_size;
  const float* X    = (const float*)d_in[0];
  const float* Wdq  = (const float*)d_in[1];
  const float* bdq  = (const float*)d_in[2];
  const float* Wdkv = (const float*)d_in[3];
  const float* bdkv = (const float*)d_in[4];
  const float* Wuq  = (const float*)d_in[5];
  const float* buq  = (const float*)d_in[6];
  const float* Wuk  = (const float*)d_in[7];
  const float* buk  = (const float*)d_in[8];
  const float* Wuv  = (const float*)d_in[9];
  const float* buv  = (const float*)d_in[10];
  const float* Wqr  = (const float*)d_in[11];
  const float* bqr  = (const float*)d_in[12];
  const float* Wkr  = (const float*)d_in[13];
  const float* bkr  = (const float*)d_in[14];
  const float* Wfc  = (const float*)d_in[15];
  const float* bfc  = (const float*)d_in[16];

  char* p = (char*)d_ws;
  auto carve = [&](size_t bytes) -> void* {
    void* r = (void*)p;
    p += (bytes + 255) & ~(size_t)255;
    return r;
  };
  auto cvt = [&](const float* s, bf16* d, long n) {
    cvt_bf16_kernel<<<(n / 4 + 255) / 256, 256, 0, stream>>>(s, d, n);
  };

  bf16* Xb    = (bf16*)carve((size_t)BSc * Dc * 2);
  bf16* Wdqb  = (bf16*)carve((size_t)Dc * DOWNc * 2);
  bf16* Wdkvb = (bf16*)carve((size_t)Dc * DOWNc * 2);
  bf16* Wuqb  = (bf16*)carve((size_t)DOWNc * UPc * 2);
  bf16* Wukb  = (bf16*)carve((size_t)DOWNc * UPc * 2);
  bf16* Wuvb  = (bf16*)carve((size_t)DOWNc * UPc * 2);
  bf16* Wqrb  = (bf16*)carve((size_t)DOWNc * Rc * Hc * 2);
  bf16* Wkrb  = (bf16*)carve((size_t)Dc * Rc * 2);
  bf16* Wfcb  = (bf16*)carve((size_t)Dc * Dc * 2);
  bf16* cqb   = (bf16*)carve((size_t)BSc * DOWNc * 2);
  bf16* ckvb  = (bf16*)carve((size_t)BSc * DOWNc * 2);
  bf16* qcb   = (bf16*)carve((size_t)BSc * UPc * 2);
  bf16* kcb   = (bf16*)carve((size_t)BSc * UPc * 2);
  bf16* vcb   = (bf16*)carve((size_t)BSc * UPc * 2);
  bf16* qrb   = (bf16*)carve((size_t)BSc * Rc * Hc * 2);
  bf16* krb   = (bf16*)carve((size_t)BSc * Rc * 2);
  bf16* Qb    = (bf16*)carve((size_t)Bc * Hc * Sc * DQK * 2);
  bf16* Kb    = (bf16*)carve((size_t)Bc * Hc * Sc * DQK * 2);
  bf16* Vtb   = (bf16*)carve((size_t)Bc * Hc * DV * Sc * 2);
  bf16* Ob    = qcb;   // qcb is dead after packing; alias for attention output

  // ---- one-time bf16 conversion of X and all weights ----
  cvt(X, Xb, (long)BSc * Dc);
  cvt(Wdq, Wdqb, (long)Dc * DOWNc);
  cvt(Wdkv, Wdkvb, (long)Dc * DOWNc);
  cvt(Wuq, Wuqb, (long)DOWNc * UPc);
  cvt(Wuk, Wukb, (long)DOWNc * UPc);
  cvt(Wuv, Wuvb, (long)DOWNc * UPc);
  cvt(Wqr, Wqrb, (long)DOWNc * Rc * Hc);
  cvt(Wkr, Wkrb, (long)Dc * Rc);
  cvt(Wfc, Wfcb, (long)Dc * Dc);

  dim3 blk256(256), blk128(128);
#define GEMM_BF16(Am, Wm, bm, Cm, M, N, K) \
  gemm_bf16_kernel<1><<<dim3((N) / 128, (M) / 128), blk256, 0, stream>>>( \
      Am, Wm, bm, nullptr, Cm, M, N, K)

  // Down projections
  GEMM_BF16(Xb, Wdqb, bdq, cqb, BSc, DOWNc, Dc);
  GEMM_BF16(Xb, Wdkvb, bdkv, ckvb, BSc, DOWNc, Dc);
  // Up projections
  GEMM_BF16(cqb, Wuqb, buq, qcb, BSc, UPc, DOWNc);
  GEMM_BF16(ckvb, Wukb, buk, kcb, BSc, UPc, DOWNc);
  GEMM_BF16(ckvb, Wuvb, buv, vcb, BSc, UPc, DOWNc);
  // Rotary projections
  GEMM_BF16(cqb, Wqrb, bqr, qrb, BSc, Rc * Hc, DOWNc);
  GEMM_BF16(Xb, Wkrb, bkr, krb, BSc, Rc, Dc);

  // Pack into attention layouts
  long tot8 = (long)BSc * UPc / 8;             // 1,048,576
  pack_heads_kernel<<<(tot8 + 255) / 256, blk256, 0, stream>>>(qcb, Qb, tot8, DQK, 0);
  pack_heads_kernel<<<(tot8 + 255) / 256, blk256, 0, stream>>>(kcb, Kb, tot8, DQK, 0);
  pack_vt_kernel<<<(tot8 + 255) / 256, blk256, 0, stream>>>(vcb, Vtb, tot8);
  long tot_qr = (long)BSc * Hc * (Rc / 2);     // 16.8M pairs
  rope_q_kernel<<<(tot_qr + 255) / 256, blk256, 0, stream>>>(qrb, Qb, tot_qr);
  long tot_kr = (long)BSc * (Rc / 2);          // 1.05M pairs
  rope_k_kernel<<<(tot_kr + 255) / 256, blk256, 0, stream>>>(krb, Kb, tot_kr);

  // Fused attention (scores stay on-chip), bf16 output
  mla_attn_kernel<<<dim3(Bc * Hc, Sc / 16), blk128, 0, stream>>>(Qb, Kb, Vtb, Ob);

  // Output projection (fp32 output to d_out)
  gemm_bf16_kernel<0><<<dim3(Dc / 128, BSc / 128), blk256, 0, stream>>>(
      Ob, Wfcb, bfc, (float*)d_out, nullptr, BSc, Dc, Dc);
#undef GEMM_BF16
}
